// SerriformNet_41120016891975
// MI455X (gfx1250) — matrix operand order repaired and checked by reading, not compiled, not run on hardware
//
#include <hip/hip_runtime.h>
#include <cstdint>
#include <cstddef>

// ---------------- problem constants (from reference) ----------------
#define B_   4
#define L_   2048
#define D_   1024
#define M_   256
#define E_   4
#define H_   256
#define KW   5
#define NTOK (B_*L_)          // 8192 tokens
#define CATW (D_+M_)          // 1280
#define EPS_ 1e-6f

typedef __attribute__((ext_vector_type(16))) _Float16 v16h;
typedef __attribute__((ext_vector_type(8)))  float    v8f;
typedef __attribute__((ext_vector_type(4)))  float    vf4;   // 16-byte copy unit
typedef __attribute__((ext_vector_type(4)))  int      v4i;

// ---- gfx1250 async Global->LDS path (ASYNCcnt), with safe fallback ----
#if __has_builtin(__builtin_amdgcn_global_load_async_to_lds_b128)
#define HAVE_ASYNC_LDS 1
#else
#define HAVE_ASYNC_LDS 0
#endif

typedef __attribute__((address_space(1))) v4i* gv4i_ptr;   // global <4 x i32>*
typedef __attribute__((address_space(3))) v4i* lv4i_ptr;   // LDS    <4 x i32>*

__device__ __forceinline__ void stage16(const _Float16* src, _Float16* dst) {
#if HAVE_ASYNC_LDS
  __builtin_amdgcn_global_load_async_to_lds_b128((gv4i_ptr)src, (lv4i_ptr)dst, 0, 0);
#else
  *(vf4*)dst = *(const vf4*)src;
#endif
}

__device__ __forceinline__ void async_wait0() {
#if HAVE_ASYNC_LDS
# if __has_builtin(__builtin_amdgcn_s_wait_asynccnt)
  __builtin_amdgcn_s_wait_asynccnt(0);
# else
  asm volatile("s_wait_asynccnt 0x0" ::: "memory");
# endif
#endif
}

// =====================================================================
//  Generic WMMA GEMM:  C[n,o] = epilogue( sum_k A[n,k] * W[o,k] )
//  A: (N x Kd) f16 row-major, leading dim lda; W: (O x Kd) f16 (ld = Kd)
//  Block tile 128x128, BK=64, double-buffered async LDS staging.
//  8 waves; each wave owns a 16x128 strip = 8 wmma accumulators.
// =====================================================================
struct GemmArgs {
  const _Float16* A; int lda;
  const _Float16* W; int Kd; int N; int O;
  const float* bias; int act;            // act: 0 none, 1 silu, 2 gelu(exact)
  const float* resid;                    // optional: += resid[n*O+o]
  const float* gates; int gstride;       // optional: *= gates[n*gstride]
  int accum;                             // Cf += v instead of =
  const float* xin; const float* rsc; const float* hres;  // final: v = x*rs + h + v
  float* Cf; _Float16* Ch; int ldch;
};

#define BM 128
#define BN 128
#define BK 64

__global__ __launch_bounds__(256) void gemm_wmma_f16(GemmArgs g) {
  __shared__ __align__(16) _Float16 As[2][BM*BK];   // 2 x 16 KB
  __shared__ __align__(16) _Float16 Bs[2][BN*BK];   // 2 x 16 KB
  const int tid  = threadIdx.x;
  const int wave = tid >> 5;
  const int lane = tid & 31;
  const int bm = blockIdx.x * BM;
  const int bn = blockIdx.y * BN;

  v8f acc[8] = {};

  // wave32 WMMA fragment lane mapping (ISA 7.12.2)
  const int mrow = wave*16 + (lane & 15);        // A: M = lane&15 within wave tile
  const int akb  = (lane < 16) ? 0 : 8;          // A: K halves per lane group
  const int nrow = lane & 15;                    // B: N = lane&15
  const int bkb  = (lane < 16) ? 0 : 8;          // B: K halves per lane group

  // stage one BMxBK A tile + BNxBK B tile into buffer `buf` (1024 chunks each)
  auto issue = [&](int buf, int k0) {
    #pragma unroll
    for (int u = 0; u < 4; u++) {
      const int c = tid + u*256;
      const int r = c >> 3, ko = (c & 7) * 8;
      stage16(g.A + (size_t)(bm + r)*g.lda + k0 + ko, &As[buf][r*BK + ko]);
    }
    #pragma unroll
    for (int u = 0; u < 4; u++) {
      const int c = tid + u*256;
      const int r = c >> 3, ko = (c & 7) * 8;
      stage16(g.W + (size_t)(bn + r)*g.Kd + k0 + ko, &Bs[buf][r*BK + ko]);
    }
  };

  issue(0, 0);
  for (int k0 = 0; k0 < g.Kd; k0 += BK) {
    const int cur = (k0 >> 6) & 1;
    async_wait0();          // own async tile landed in LDS
    __syncthreads();        // everyone's tile published; prev reads of cur^1 done
    if (k0 + BK < g.Kd) issue(cur ^ 1, k0 + BK);

    #pragma unroll
    for (int kk = 0; kk < 2; kk++) {          // two 32-wide K slices
      // ---- load phase: A fragment + all 8 B fragments (distinct regs,
      //      many ds_load_b128 in flight -> latency amortized) ----
      v16h afrag;
      const _Float16* ap = &As[cur][mrow*BK + kk*32 + akb];
      *((vf4*)&afrag + 0) = *(const vf4*)(ap);
      *((vf4*)&afrag + 1) = *(const vf4*)(ap + 16);
      v16h bfr[8];
      #pragma unroll
      for (int nt = 0; nt < 8; nt++) {
        const _Float16* bp = &Bs[cur][(nt*16 + nrow)*BK + kk*32 + bkb];
        *((vf4*)&bfr[nt] + 0) = *(const vf4*)(bp);
        *((vf4*)&bfr[nt] + 1) = *(const vf4*)(bp + 16);
      }
      // ---- compute phase: 8 back-to-back WMMAs ----
      #pragma unroll
      for (int nt = 0; nt < 8; nt++) {
        acc[nt] = __builtin_amdgcn_wmma_f32_16x16x32_f16(
            /*neg_a=*/false, afrag, /*neg_b=*/false, bfr[nt],
            /*c_mod=*/(short)0, acc[nt], /*reuse_a=*/false, /*reuse_b=*/false);
      }
    }
  }

  // ---- epilogue: C/D layout: VGPR r -> M = r (lanes 0-15) / r+8 (lanes 16-31)
  const int colbase = bn + (lane & 15);
  const int rbase   = bm + wave*16 + ((lane < 16) ? 0 : 8);
  #pragma unroll
  for (int nt = 0; nt < 8; nt++) {
    const int col = colbase + nt*16;
    #pragma unroll
    for (int r = 0; r < 8; r++) {
      const int row = rbase + r;
      float v = acc[nt][r];
      const size_t idx = (size_t)row * g.O + col;
      if (g.bias) v += g.bias[col];
      if (g.act == 1)      v = v / (1.f + __expf(-v));             // silu
      else if (g.act == 2) v = 0.5f*v*(1.f + erff(v*0.70710678f)); // gelu exact
      if (g.gates) v *= g.gates[(size_t)row * g.gstride];
      if (g.resid) v += g.resid[idx];
      if (g.xin)   v = g.xin[idx]*g.rsc[0] + g.hres[idx] + v;
      if (g.Cf) { if (g.accum) g.Cf[idx] += v; else g.Cf[idx] = v; }
      if (g.Ch) g.Ch[(size_t)row * g.ldch + col] = (_Float16)v;
    }
  }
}

// =====================================================================
//  Support kernels
// =====================================================================
__global__ void k_f32_to_f16(const float* __restrict__ s, _Float16* __restrict__ d, int n) {
  int i = blockIdx.x*256 + threadIdx.x;
  if (i < n) d[i] = (_Float16)s[i];
}

// per-token rstd of x (first RMSNorm, weight folded into conv kernel)
__global__ __launch_bounds__(256) void k_rms_rstd(const float* __restrict__ x,
                                                  float* __restrict__ rstd) {
  __shared__ float red[256];
  const int n = blockIdx.x;
  const float* xr = x + (size_t)n*D_;
  float s = 0.f;
  for (int d = threadIdx.x; d < D_; d += 256) { float v = xr[d]; s += v*v; }
  red[threadIdx.x] = s; __syncthreads();
  for (int w = 128; w > 0; w >>= 1) {
    if (threadIdx.x < w) red[threadIdx.x] += red[threadIdx.x+w];
    __syncthreads();
  }
  if (threadIdx.x == 0) rstd[n] = rsqrtf(red[0]/(float)D_ + EPS_);
}

// fused RMSNorm + causal depthwise conv (K=5) -> f16 for pw GEMM
__global__ __launch_bounds__(256) void k_conv(const float* __restrict__ x,
    const float* __restrict__ rstd, const float* __restrict__ nw,
    const float* __restrict__ dww, const float* __restrict__ dwb,
    _Float16* __restrict__ y16) {
  const int i = blockIdx.x*256 + threadIdx.x;    // over NTOK*D
  const int d = i & (D_-1);
  const int n = i >> 10;
  const int b = n / L_, l = n % L_;
  float acc = dwb[d];
  const float wn = nw[d];
  #pragma unroll
  for (int k = 0; k < KW; k++) {
    const int t = l - (KW-1) + k;
    if (t >= 0) {
      acc += dww[d*KW + k] * x[((size_t)b*L_ + t)*D_ + d] * rstd[b*L_ + t] * wn;
    }
  }
  y16[i] = (_Float16)acc;
}

// recurrent decay scan: one lane per (b,m) chain; writes wv (f16) into
// cat buffer cols [1024,1280) and next_mem (f32) at the tail of d_out.
__global__ void k_scan(const float* __restrict__ vf, const float* __restrict__ mem,
                       const float* __restrict__ td, const float* __restrict__ tf,
                       _Float16* __restrict__ cat16, float* __restrict__ nextmem) {
  const int i = blockIdx.x*blockDim.x + threadIdx.x;
  if (i >= B_*M_) return;
  const int b = i / M_, m = i % M_;
  const float decay  = 0.9f/(1.f + __expf(-td[m])) + 0.1f;
  const float tfirst = 1.f /(1.f + __expf(-tf[m]));
  const float base = mem[b*M_ + m] * tfirst;
  float w = 0.f;
  for (int t = 0; t < L_; t++) {
    w = w*decay + vf[((size_t)b*L_ + t)*M_ + m];
    const float wv = w + base;
    cat16[((size_t)b*L_ + t)*CATW + D_ + m] = (_Float16)wv;
    if (t == L_-1) nextmem[b*M_ + m] = wv;
  }
}

// router: logits -> top-2 softmax -> dense gates (NTOK x E)
__global__ __launch_bounds__(128) void k_router(const float* __restrict__ h3,
    const float* __restrict__ rw, const float* __restrict__ rb,
    float* __restrict__ gates) {
  __shared__ float red[128*E_];
  const int n = blockIdx.x;
  const float* hr = h3 + (size_t)n*D_;
  float acc[E_] = {0.f, 0.f, 0.f, 0.f};
  for (int d = threadIdx.x; d < D_; d += 128) {
    const float v = hr[d];
    #pragma unroll
    for (int e = 0; e < E_; e++) acc[e] += v * rw[e*D_ + d];
  }
  #pragma unroll
  for (int e = 0; e < E_; e++) red[e*128 + threadIdx.x] = acc[e];
  __syncthreads();
  for (int w = 64; w > 0; w >>= 1) {
    if (threadIdx.x < w)
      #pragma unroll
      for (int e = 0; e < E_; e++) red[e*128+threadIdx.x] += red[e*128+threadIdx.x+w];
    __syncthreads();
  }
  if (threadIdx.x == 0) {
    float lg[E_];
    #pragma unroll
    for (int e = 0; e < E_; e++) lg[e] = red[e*128] + rb[e];
    int i0 = 0;
    for (int e = 1; e < E_; e++) if (lg[e] > lg[i0]) i0 = e;
    int i1 = -1;
    for (int e = 0; e < E_; e++) if (e != i0 && (i1 < 0 || lg[e] > lg[i1])) i1 = e;
    const float e1 = __expf(lg[i1] - lg[i0]);
    const float w0 = 1.f/(1.f + e1), w1 = e1/(1.f + e1);
    #pragma unroll
    for (int e = 0; e < E_; e++) gates[(size_t)n*E_ + e] = 0.f;
    gates[(size_t)n*E_ + i0] = w0;
    gates[(size_t)n*E_ + i1] = w1;
  }
}

// second RMSNorm (fnorm_w), emits f32 + f16
__global__ __launch_bounds__(256) void k_rmsnorm(const float* __restrict__ t,
    const float* __restrict__ w, float* __restrict__ of, _Float16* __restrict__ oh) {
  __shared__ float red[256];
  const int n = blockIdx.x;
  const float* tr = t + (size_t)n*D_;
  float v[4]; float s = 0.f;
  #pragma unroll
  for (int j = 0; j < 4; j++) { v[j] = tr[threadIdx.x + j*256]; s += v[j]*v[j]; }
  red[threadIdx.x] = s; __syncthreads();
  for (int ww = 128; ww > 0; ww >>= 1) {
    if (threadIdx.x < ww) red[threadIdx.x] += red[threadIdx.x+ww];
    __syncthreads();
  }
  const float r = rsqrtf(red[0]/(float)D_ + EPS_);
  #pragma unroll
  for (int j = 0; j < 4; j++) {
    const int d = threadIdx.x + j*256;
    const float o = w[d]*v[j]*r;
    of[(size_t)n*D_ + d] = o;
    oh[(size_t)n*D_ + d] = (_Float16)o;
  }
}

// =====================================================================
//  Host orchestration
// =====================================================================
extern "C" void kernel_launch(void* const* d_in, const int* in_sizes, int n_in,
                              void* d_out, int out_size, void* d_ws, size_t ws_size,
                              hipStream_t stream) {
  const float* x       = (const float*)d_in[0];
  const float* mem     = (const float*)d_in[1];
  const float* norm_w  = (const float*)d_in[2];
  const float* dw_w    = (const float*)d_in[3];
  const float* dw_b    = (const float*)d_in[4];
  const float* pw_w    = (const float*)d_in[5];
  const float* pw_b    = (const float*)d_in[6];
  const float* val_w   = (const float*)d_in[7];
  const float* val_b   = (const float*)d_in[8];
  const float* tdec    = (const float*)d_in[9];
  const float* tfst    = (const float*)d_in[10];
  const float* gate_w  = (const float*)d_in[11];
  const float* gate_b  = (const float*)d_in[12];
  const float* rt_w    = (const float*)d_in[13];
  const float* rt_b    = (const float*)d_in[14];
  const float* exp_w   = (const float*)d_in[15];
  const float* exp_b   = (const float*)d_in[16];
  const float* out_w   = (const float*)d_in[17];
  const float* out_b   = (const float*)d_in[18];
  const float* fnorm_w = (const float*)d_in[19];
  const float* down_w  = (const float*)d_in[20];
  const float* down_b  = (const float*)d_in[21];
  const float* up_w    = (const float*)d_in[22];
  const float* up_b    = (const float*)d_in[23];
  const float* res_sc  = (const float*)d_in[24];

  float* outp    = (float*)d_out;                 // (B,L,D) f32
  float* nextmem = outp + (size_t)NTOK*D_;        // (B,M)  f32

  // ---- workspace bump allocator ----
  char* ws = (char*)d_ws; size_t off = 0;
  auto alloc = [&](size_t bytes) -> void* {
    void* p = ws + off; off = (off + bytes + 255) & ~(size_t)255; return p;
  };
  _Float16* pw16   = (_Float16*)alloc((size_t)D_*D_*2);
  _Float16* val16  = (_Float16*)alloc((size_t)M_*D_*2);
  _Float16* gate16 = (_Float16*)alloc((size_t)D_*CATW*2);
  _Float16* exp16  = (_Float16*)alloc((size_t)E_*D_*D_*2);
  _Float16* outw16 = (_Float16*)alloc((size_t)D_*D_*2);
  _Float16* down16 = (_Float16*)alloc((size_t)H_*D_*2);
  _Float16* up16   = (_Float16*)alloc((size_t)D_*H_*2);
  float*    rstd   = (float*)   alloc((size_t)NTOK*4);
  _Float16* y16    = (_Float16*)alloc((size_t)NTOK*D_*2);
  float*    h2f    = (float*)   alloc((size_t)NTOK*D_*4);
  _Float16* cat16  = (_Float16*)alloc((size_t)NTOK*CATW*2);
  float*    vf     = (float*)   alloc((size_t)NTOK*M_*4);
  float*    h3f    = (float*)   alloc((size_t)NTOK*D_*4);
  _Float16* h3h    = (_Float16*)alloc((size_t)NTOK*D_*2);
  float*    gatesb = (float*)   alloc((size_t)NTOK*E_*4);
  float*    comb   = (float*)   alloc((size_t)NTOK*D_*4);
  _Float16* comb16 = (_Float16*)alloc((size_t)NTOK*D_*2);
  float*    t4f    = (float*)   alloc((size_t)NTOK*D_*4);
  float*    h4f    = (float*)   alloc((size_t)NTOK*D_*4);
  _Float16* h4h    = (_Float16*)alloc((size_t)NTOK*D_*2);
  _Float16* ff16   = (_Float16*)alloc((size_t)NTOK*H_*2);
  (void)ws_size; (void)in_sizes; (void)n_in; (void)out_size;

  auto cvt = [&](const float* s, _Float16* d, int n) {
    k_f32_to_f16<<<(n + 255)/256, 256, 0, stream>>>(s, d, n);
  };
  auto gemm = [&](const _Float16* A, int lda, const _Float16* W, int Kd, int O,
                  const float* bias, int act, const float* resid,
                  const float* gates, int gstride, int accum,
                  const float* xin, const float* rsc, const float* hres,
                  float* Cf, _Float16* Ch, int ldch) {
    GemmArgs a{A, lda, W, Kd, NTOK, O, bias, act, resid,
               gates, gstride, accum, xin, rsc, hres, Cf, Ch, ldch};
    dim3 grid(NTOK/BM, O/BN);
    gemm_wmma_f16<<<grid, 256, 0, stream>>>(a);
  };

  // 1) weights -> f16
  cvt(pw_w,   pw16,   D_*D_);
  cvt(val_w,  val16,  M_*D_);
  cvt(gate_w, gate16, D_*CATW);
  cvt(exp_w,  exp16,  E_*D_*D_);
  cvt(out_w,  outw16, D_*D_);
  cvt(down_w, down16, H_*D_);
  cvt(up_w,   up16,   D_*H_);

  // 2) RMSNorm rstd + fused norm/depthwise-conv -> y16
  k_rms_rstd<<<NTOK, 256, 0, stream>>>(x, rstd);
  k_conv<<<(NTOK*D_)/256, 256, 0, stream>>>(x, rstd, norm_w, dw_w, dw_b, y16);

  // 3) pointwise: h2 = silu(y @ pw^T + b); f16 copy into cat[:, :1024]
  gemm(y16, D_, pw16, D_, D_, pw_b, /*silu*/1, nullptr, nullptr, 0, 0,
       nullptr, nullptr, nullptr, h2f, cat16, CATW);

  // 4) value projection: v = h2 @ val^T + b   (reads cat cols [0,1024))
  gemm(cat16, CATW, val16, D_, M_, val_b, 0, nullptr, nullptr, 0, 0,
       nullptr, nullptr, nullptr, vf, nullptr, 0);

  // 5) recurrent decay scan -> cat[:, 1024:1280) + next_mem
  k_scan<<<(B_*M_ + 255)/256, 256, 0, stream>>>(vf, mem, tdec, tfst, cat16, nextmem);

  // 6) gate: h3 = h2 + [h2, wv] @ gate^T + b    (f32 + f16)
  gemm(cat16, CATW, gate16, CATW, D_, gate_b, 0, h2f, nullptr, 0, 0,
       nullptr, nullptr, nullptr, h3f, h3h, D_);

  // 7) router top-2 softmax gates
  k_router<<<NTOK, 128, 0, stream>>>(h3f, rt_w, rt_b, gatesb);

  // 8) experts (dense, gated accumulate into comb)
  (void)hipMemsetAsync(comb, 0, (size_t)NTOK*D_*4, stream);
  for (int e = 0; e < E_; e++) {
    gemm(h3h, D_, exp16 + (size_t)e*D_*D_, D_, D_, exp_b + (size_t)e*D_, /*silu*/1,
         nullptr, gatesb + e, E_, /*accum*/1, nullptr, nullptr, nullptr,
         comb, nullptr, 0);
  }
  cvt(comb, comb16, NTOK*D_);

  // 9) out projection + residual, then RMSNorm (fnorm)
  gemm(comb16, D_, outw16, D_, D_, out_b, 0, h3f, nullptr, 0, 0,
       nullptr, nullptr, nullptr, t4f, nullptr, 0);
  k_rmsnorm<<<NTOK, 256, 0, stream>>>(t4f, fnorm_w, h4f, h4h);

  // 10) FFN: down (gelu exact) -> up with final residual epilogue
  gemm(h4h, D_, down16, D_, H_, down_b, /*gelu*/2, nullptr, nullptr, 0, 0,
       nullptr, nullptr, nullptr, nullptr, ff16, H_);
  gemm(ff16, H_, up16, H_, D_, up_b, 0, nullptr, nullptr, 0, 0,
       x, res_sc, h4f, outp, nullptr, 0);
}
